// GPT2AttentionKIVI_64561948393474
// MI455X (gfx1250) — matrix operand order repaired and checked by reading, not compiled, not run on hardware
//
#include <hip/hip_runtime.h>

// ---------------------------------------------------------------------------
// GPT2 attention with KIVI 2-bit KV fake-quant, CDNA5 (gfx1250) WMMA bf16 path
// B=2, S=2048, E=1024, H=16, D=64, GROUP=64, RESIDUAL=32, K_BITS=V_BITS=2
// ---------------------------------------------------------------------------

#define B_ 2
#define S_ 2048
#define E_ 1024
#define H_ 16
#define D_ 64
#define E3_ 3072
#define GRP_ 64
#define RES_ 32
#define CLIP_ 10000.0f

typedef __attribute__((ext_vector_type(16))) __bf16 v16bf;
typedef __attribute__((ext_vector_type(8)))  float  v8f;

union FragBF {
    uint4 u4[2];
    v16bf v;
};

__device__ __forceinline__ unsigned short f32_to_bf16_bits(float f) {
    union { float f; unsigned int u; } a;
    a.f = f;
    unsigned int r = a.u + 0x7FFFu + ((a.u >> 16) & 1u);  // RTNE
    return (unsigned short)(r >> 16);
}

// ---------------------------------------------------------------------------
// elementwise cast f32 -> bf16 bits
// ---------------------------------------------------------------------------
__global__ void cast_bf16_kernel(const float* __restrict__ src,
                                 unsigned short* __restrict__ dst, int n) {
    int i = blockIdx.x * blockDim.x + threadIdx.x;
    if (i < n) dst[i] = f32_to_bf16_bits(src[i]);
}

// W [K,N] f32 -> Wt [N,K] bf16  (so GEMM B-fragments are contiguous along K)
__global__ void transpose_cast_kernel(const float* __restrict__ W,
                                      unsigned short* __restrict__ Wt,
                                      int K, int N) {
    int i = blockIdx.x * blockDim.x + threadIdx.x;
    if (i >= K * N) return;
    int n = i / K, k = i - n * K;
    Wt[(size_t)n * K + k] = f32_to_bf16_bits(W[(size_t)k * N + n]);
}

// ---------------------------------------------------------------------------
// bf16 WMMA GEMM: C[M,N] = A[M,K] * Bt[N,K]^T + bias
// one wave -> 32(M) x 64(N) output block, double-buffered over K
// ---------------------------------------------------------------------------
__device__ __forceinline__ void load_frags(
    const unsigned short* arow0, const unsigned short* arow1,
    const unsigned short* brow0, const unsigned short* brow1,
    const unsigned short* brow2, const unsigned short* brow3,
    int k0, FragBF a[2], FragBF b[4]) {
    a[0].u4[0] = *(const uint4*)(arow0 + k0);
    a[0].u4[1] = *(const uint4*)(arow0 + k0 + 16);
    a[1].u4[0] = *(const uint4*)(arow1 + k0);
    a[1].u4[1] = *(const uint4*)(arow1 + k0 + 16);
    b[0].u4[0] = *(const uint4*)(brow0 + k0);
    b[0].u4[1] = *(const uint4*)(brow0 + k0 + 16);
    b[1].u4[0] = *(const uint4*)(brow1 + k0);
    b[1].u4[1] = *(const uint4*)(brow1 + k0 + 16);
    b[2].u4[0] = *(const uint4*)(brow2 + k0);
    b[2].u4[1] = *(const uint4*)(brow2 + k0 + 16);
    b[3].u4[0] = *(const uint4*)(brow3 + k0);
    b[3].u4[1] = *(const uint4*)(brow3 + k0 + 16);
}

__device__ __forceinline__ void wmma8(v8f acc[2][4], FragBF a[2], FragBF b[4]) {
#pragma unroll
    for (int mt = 0; mt < 2; ++mt)
#pragma unroll
        for (int t = 0; t < 4; ++t)
            acc[mt][t] = __builtin_amdgcn_wmma_f32_16x16x32_bf16(
                false, a[mt].v, false, b[t].v, (short)0, acc[mt][t], false,
                false);
}

__global__ __launch_bounds__(256) void gemm_bf16_wmma_kernel(
    const unsigned short* __restrict__ A,   // [M,K] bf16 bits
    const unsigned short* __restrict__ Bt,  // [N,K] bf16 bits
    const float* __restrict__ bias,         // [N]
    float* __restrict__ C,                  // [M,N]
    int M, int N, int K) {
    int lane  = threadIdx.x & 31;
    int wave  = (blockIdx.x * blockDim.x + threadIdx.x) >> 5;
    int tilesN = N >> 6;
    int tm = wave / tilesN;
    int tn = wave - tm * tilesN;
    if (tm * 32 >= M) return;
    int m0 = tm * 32, n0 = tn * 64;
    int l16 = lane & 15;
    int hi  = (lane < 16) ? 0 : 1;

    // A fragment rows (two 16-row M tiles), per ISA A layout
    const unsigned short* arow0 = A + (size_t)(m0 + l16) * K + hi * 8;
    const unsigned short* arow1 = arow0 + (size_t)16 * K;
    // B fragment rows (four 16-col N tiles), per ISA B layout
    const unsigned short* brow0 = Bt + (size_t)(n0 + l16) * K + hi * 8;
    const unsigned short* brow1 = brow0 + (size_t)16 * K;
    const unsigned short* brow2 = brow0 + (size_t)32 * K;
    const unsigned short* brow3 = brow0 + (size_t)48 * K;

    v8f acc[2][4] = {};
    FragBF aP[2], bP[4];  // ping
    FragBF aQ[2], bQ[4];  // pong

    load_frags(arow0, arow1, brow0, brow1, brow2, brow3, 0, aP, bP);

    // K multiple of 64 (K=1024 here); two k-steps per iteration, ping-pong
    for (int k0 = 0; k0 < K; k0 += 64) {
        // prefetch 2 iterations ahead on the B stream
        int kpf = k0 + 128;
        if (kpf < K) {
            __builtin_prefetch(brow0 + kpf, 0, 1);
            __builtin_prefetch(brow1 + kpf, 0, 1);
            __builtin_prefetch(brow2 + kpf, 0, 1);
            __builtin_prefetch(brow3 + kpf, 0, 1);
        }
        // issue loads for step k0+32 before consuming step k0
        load_frags(arow0, arow1, brow0, brow1, brow2, brow3, k0 + 32, aQ, bQ);
        wmma8(acc, aP, bP);
        // issue loads for step k0+64 (clamped on the final iteration)
        int k2 = (k0 + 64 < K) ? (k0 + 64) : 0;
        load_frags(arow0, arow1, brow0, brow1, brow2, brow3, k2, aP, bP);
        wmma8(acc, aQ, bQ);
    }

#pragma unroll
    for (int mt = 0; mt < 2; ++mt) {
#pragma unroll
        for (int t = 0; t < 4; ++t) {
            int col = n0 + 16 * t + l16;
            float bv = bias ? bias[col] : 0.0f;
#pragma unroll
            for (int j = 0; j < 8; ++j) {
                int row = m0 + 16 * mt + j + 8 * hi;
                C[(size_t)row * N + col] = acc[mt][t][j] + bv;
            }
        }
    }
}

// ---------------------------------------------------------------------------
// split Q out of qkv (f32) into bf16 [B,H,S,D]
// ---------------------------------------------------------------------------
__global__ void qsplit_kernel(const float* __restrict__ qkv,
                              unsigned short* __restrict__ Q) {
    int i = blockIdx.x * blockDim.x + threadIdx.x;
    if (i >= B_ * H_ * S_ * D_) return;
    int d  = i & (D_ - 1);
    int s  = (i >> 6) & (S_ - 1);
    int bh = i >> 17;
    int h  = bh & (H_ - 1);
    int b  = bh >> 4;
    Q[i] = f32_to_bf16_bits(qkv[(size_t)(b * S_ + s) * E3_ + h * D_ + d]);
}

// ---------------------------------------------------------------------------
// KIVI key quantization: per-channel min/max over 64-token groups (2 bits)
// all S tokens are quantized (kql == S). Output Kq bf16 [B,H,S,D].
// one thread = one (b,h,group,channel)
// ---------------------------------------------------------------------------
__global__ void quantK_kernel(const float* __restrict__ qkv,
                              unsigned short* __restrict__ Kq) {
    int i = blockIdx.x * blockDim.x + threadIdx.x;
    if (i >= B_ * H_ * (S_ / GRP_) * D_) return;
    int d  = i & (D_ - 1);
    int g  = (i >> 6) & ((S_ / GRP_) - 1);
    int bh = i >> 11;
    int h  = bh & (H_ - 1);
    int b  = bh >> 4;

    const float* base =
        qkv + (size_t)(b * S_ + g * GRP_) * E3_ + E_ + h * D_ + d;
    float vals[GRP_];
    float mn = 1e30f, mx = -1e30f;
#pragma unroll 8
    for (int t = 0; t < GRP_; ++t) {
        float v = base[(size_t)t * E3_];
        v = fminf(fmaxf(v, -CLIP_), CLIP_);
        vals[t] = v;
        mn = fminf(mn, v);
        mx = fmaxf(mx, v);
    }
    float scale = fmaxf((mx - mn) * (1.0f / 3.0f), 1e-8f);  // 2 bits -> qmax=3
    float inv = 1.0f / scale;
    unsigned short* out = Kq + ((size_t)bh * S_ + g * GRP_) * D_ + d;
#pragma unroll 8
    for (int t = 0; t < GRP_; ++t) {
        float q   = roundf((vals[t] - mn) * inv);
        float deq = q * scale + mn;
        out[(size_t)t * D_] = f32_to_bf16_bits(deq);
    }
}

// ---------------------------------------------------------------------------
// KIVI value quantization: per-token min/max over the 64-channel group,
// last RESIDUAL tokens copied raw. Output transposed Vt bf16 [B,H,D,S].
// one thread = one (b,h,token)
// ---------------------------------------------------------------------------
__global__ void quantV_kernel(const float* __restrict__ qkv,
                              unsigned short* __restrict__ Vt) {
    int i = blockIdx.x * blockDim.x + threadIdx.x;
    if (i >= B_ * H_ * S_) return;
    int s  = i & (S_ - 1);
    int bh = i >> 11;
    int h  = bh & (H_ - 1);
    int b  = bh >> 4;

    const float* base = qkv + (size_t)(b * S_ + s) * E3_ + 2 * E_ + h * D_;
    unsigned short* out = Vt + (size_t)bh * D_ * S_ + s;

    if (s < S_ - RES_) {
        float vals[D_];
        float mn = 1e30f, mx = -1e30f;
#pragma unroll 8
        for (int d = 0; d < D_; ++d) {
            float v = base[d];
            v = fminf(fmaxf(v, -CLIP_), CLIP_);
            vals[d] = v;
            mn = fminf(mn, v);
            mx = fmaxf(mx, v);
        }
        float scale = fmaxf((mx - mn) * (1.0f / 3.0f), 1e-8f);
        float inv = 1.0f / scale;
#pragma unroll 8
        for (int d = 0; d < D_; ++d) {
            float q   = roundf((vals[d] - mn) * inv);
            float deq = q * scale + mn;
            out[(size_t)d * S_] = f32_to_bf16_bits(deq);
        }
    } else {
#pragma unroll 8
        for (int d = 0; d < D_; ++d)
            out[(size_t)d * S_] = f32_to_bf16_bits(base[d]);
    }
}

// ---------------------------------------------------------------------------
// flash attention, one wave (32 threads) per (b, h, 16-row q tile)
// Q bf16 [B,H,S,D] (A layout), Kq bf16 [B,H,S,D] (Bt layout for QK^T),
// Vt bf16 [B,H,D,S] (Bt layout for P*V). Output Y bf16 [B,S,H*D].
// ---------------------------------------------------------------------------
__global__ __launch_bounds__(32) void attn_kernel(
    const unsigned short* __restrict__ Q,
    const unsigned short* __restrict__ Kq,
    const unsigned short* __restrict__ Vt,
    unsigned short* __restrict__ Y) {
    __shared__ unsigned short pbuf[16 * 32];

    int lane = threadIdx.x;
    int qt = blockIdx.x & 127;
    int bh = blockIdx.x >> 7;
    int h  = bh & (H_ - 1);
    int b  = bh >> 4;
    int q0 = qt * 16;
    int l16 = lane & 15;
    int hi  = (lane < 16) ? 0 : 1;

    const float sc = 0.125f * 1.44269504089f;  // 1/sqrt(64) * log2(e)

    // Q A-fragments for the two 32-wide k-steps over D=64 (held all loop)
    const unsigned short* qrow =
        Q + ((size_t)bh * S_ + q0 + l16) * D_ + hi * 8;
    FragBF qa0, qa1;
    qa0.u4[0] = *(const uint4*)(qrow);
    qa0.u4[1] = *(const uint4*)(qrow + 16);
    qa1.u4[0] = *(const uint4*)(qrow + 32);
    qa1.u4[1] = *(const uint4*)(qrow + 48);

    float m[8], l[8];
    v8f o[4] = {};
#pragma unroll
    for (int j = 0; j < 8; ++j) { m[j] = -1e30f; l[j] = 0.0f; }

    for (int j0 = 0; j0 <= q0; j0 += 32) {
        // ---- S = Q K^T over a 16(q) x 32(keys) block, in log2 domain ----
        v8f s[2] = {};
#pragma unroll
        for (int t = 0; t < 2; ++t) {
            const unsigned short* kp =
                Kq + ((size_t)bh * S_ + j0 + 16 * t + l16) * D_ + hi * 8;
            FragBF kb0, kb1;
            kb0.u4[0] = *(const uint4*)(kp);
            kb0.u4[1] = *(const uint4*)(kp + 16);
            kb1.u4[0] = *(const uint4*)(kp + 32);
            kb1.u4[1] = *(const uint4*)(kp + 48);
            s[t] = __builtin_amdgcn_wmma_f32_16x16x32_bf16(
                false, qa0.v, false, kb0.v, (short)0, s[t], false, false);
            s[t] = __builtin_amdgcn_wmma_f32_16x16x32_bf16(
                false, qa1.v, false, kb1.v, (short)0, s[t], false, false);
        }
        // ---- scale + causal mask ----
#pragma unroll
        for (int t = 0; t < 2; ++t) {
            int col = j0 + 16 * t + l16;
#pragma unroll
            for (int j = 0; j < 8; ++j) {
                int row = q0 + j + 8 * hi;
                float v = s[t][j] * sc;
                s[t][j] = (col > row) ? -1e30f : v;
            }
        }
        // ---- online softmax: row max across 32 cols (16-lane groups) ----
        float alpha[8];
#pragma unroll
        for (int j = 0; j < 8; ++j) {
            float v = fmaxf(s[0][j], s[1][j]);
#pragma unroll
            for (int off = 1; off < 16; off <<= 1)
                v = fmaxf(v, __shfl_xor(v, off, 32));
            float mn = fmaxf(m[j], v);
            alpha[j] = exp2f(m[j] - mn);
            m[j] = mn;
        }
#pragma unroll
        for (int t = 0; t < 4; ++t)
#pragma unroll
            for (int j = 0; j < 8; ++j) o[t][j] *= alpha[j];
        // ---- p = exp2(s - m), row sums ----
#pragma unroll
        for (int t = 0; t < 2; ++t)
#pragma unroll
            for (int j = 0; j < 8; ++j) s[t][j] = exp2f(s[t][j] - m[j]);
#pragma unroll
        for (int j = 0; j < 8; ++j) {
            float r = s[0][j] + s[1][j];
#pragma unroll
            for (int off = 1; off < 16; off <<= 1)
                r += __shfl_xor(r, off, 32);
            l[j] = l[j] * alpha[j] + r;
        }
        // ---- C-layout -> A-layout via LDS (bf16) ----
        __syncthreads();
#pragma unroll
        for (int t = 0; t < 2; ++t)
#pragma unroll
            for (int j = 0; j < 8; ++j)
                pbuf[(j + 8 * hi) * 32 + 16 * t + l16] =
                    f32_to_bf16_bits(s[t][j]);
        __syncthreads();
        FragBF pa;
        const unsigned short* pp = pbuf + l16 * 32 + hi * 8;
        pa.u4[0] = *(const uint4*)(pp);
        pa.u4[1] = *(const uint4*)(pp + 16);
        // ---- O += P * V over D=64 (4 N-tiles) ----
#pragma unroll
        for (int t2 = 0; t2 < 4; ++t2) {
            const unsigned short* vp =
                Vt + ((size_t)bh * D_ + 16 * t2 + l16) * S_ + j0 + hi * 8;
            FragBF vb;
            vb.u4[0] = *(const uint4*)(vp);
            vb.u4[1] = *(const uint4*)(vp + 16);
            o[t2] = __builtin_amdgcn_wmma_f32_16x16x32_bf16(
                false, pa.v, false, vb.v, (short)0, o[t2], false, false);
        }
    }
    // ---- normalize and write Y[B,S,H*D] bf16 ----
    float inv[8];
#pragma unroll
    for (int j = 0; j < 8; ++j) inv[j] = 1.0f / l[j];
#pragma unroll
    for (int t2 = 0; t2 < 4; ++t2) {
#pragma unroll
        for (int j = 0; j < 8; ++j) {
            int row = q0 + j + 8 * hi;
            Y[(size_t)(b * S_ + row) * E_ + h * D_ + 16 * t2 + l16] =
                f32_to_bf16_bits(o[t2][j] * inv[j]);
        }
    }
}

// ---------------------------------------------------------------------------
// launch
// ---------------------------------------------------------------------------
extern "C" void kernel_launch(void* const* d_in, const int* in_sizes, int n_in,
                              void* d_out, int out_size, void* d_ws,
                              size_t ws_size, hipStream_t stream) {
    const float* x      = (const float*)d_in[0];  // [B,S,E]
    const float* W_qkv  = (const float*)d_in[1];  // [E,3E]
    const float* b_qkv  = (const float*)d_in[2];  // [3E]
    const float* W_proj = (const float*)d_in[3];  // [E,E]
    const float* b_proj = (const float*)d_in[4];  // [E]
    float* out = (float*)d_out;                   // [B,S,E]

    const int M = B_ * S_;  // 4096
    // workspace carve-up (bf16 = unsigned short)
    unsigned short* Xb     = (unsigned short*)d_ws;              // M*E
    unsigned short* WqkvT  = Xb    + (size_t)M * E_;             // 3E*E
    unsigned short* WprojT = WqkvT + (size_t)E3_ * E_;           // E*E
    unsigned short* Qb     = WprojT + (size_t)E_ * E_;           // B*H*S*D
    unsigned short* Kqb    = Qb  + (size_t)B_ * H_ * S_ * D_;
    unsigned short* Vtb    = Kqb + (size_t)B_ * H_ * S_ * D_;
    unsigned short* Yb     = Vtb + (size_t)B_ * H_ * S_ * D_;    // M*E
    float* qkv = (float*)(Yb + (size_t)M * E_);                  // M*3E f32

    // 1. casts
    {
        int n = M * E_;
        cast_bf16_kernel<<<(n + 255) / 256, 256, 0, stream>>>(x, Xb, n);
    }
    {
        int n = E_ * E3_;
        transpose_cast_kernel<<<(n + 255) / 256, 256, 0, stream>>>(
            W_qkv, WqkvT, E_, E3_);
    }
    {
        int n = E_ * E_;
        transpose_cast_kernel<<<(n + 255) / 256, 256, 0, stream>>>(
            W_proj, WprojT, E_, E_);
    }
    // 2. QKV GEMM (f32 out + bias): (M/32)*(N/64) waves, 8 waves/block
    {
        int waves = (M / 32) * (E3_ / 64);
        gemm_bf16_wmma_kernel<<<waves / 8, 256, 0, stream>>>(
            Xb, WqkvT, b_qkv, qkv, M, E3_, E_);
    }
    // 3. Q split, K/V KIVI quantization
    {
        int n = B_ * H_ * S_ * D_;
        qsplit_kernel<<<(n + 255) / 256, 256, 0, stream>>>(qkv, Qb);
    }
    {
        int n = B_ * H_ * (S_ / GRP_) * D_;
        quantK_kernel<<<(n + 255) / 256, 256, 0, stream>>>(qkv, Kqb);
    }
    {
        int n = B_ * H_ * S_;
        quantV_kernel<<<(n + 255) / 256, 256, 0, stream>>>(qkv, Vtb);
    }
    // 4. flash attention
    {
        int blocks = B_ * H_ * (S_ / 16);
        attn_kernel<<<blocks, 32, 0, stream>>>(Qb, Kqb, Vtb, Yb);
    }
    // 5. output projection -> d_out (f32)
    {
        int waves = (M / 32) * (E_ / 64);
        gemm_bf16_wmma_kernel<<<waves / 8, 256, 0, stream>>>(
            Yb, WprojT, b_proj, out, M, E_, E_);
    }
}
